// BertSelfAttention_81750407512311
// MI455X (gfx1250) — matrix operand order repaired and checked by reading, not compiled, not run on hardware
//
#include <hip/hip_runtime.h>
#include <hip/hip_bf16.h>
#include <stdint.h>

#define B_  8
#define S_  1024
#define H_  1024
#define NH_ 16
#define HD_ 64

typedef __attribute__((ext_vector_type(16))) __bf16 v16bf;
typedef __attribute__((ext_vector_type(8)))  float  v8f;
typedef unsigned short u16;

#define WSTRIDE 40   // padded LDS row stride (u16 units): 80B -> 16 distinct banks, 16B aligned

__device__ __forceinline__ u16 f32_to_bf16(float f) {
    union { float f; uint32_t u; } x; x.f = f;
    uint32_t u = x.u;
    return (u16)((u + 0x7FFFu + ((u >> 16) & 1u)) >> 16);   // RNE
}

// ---------------------------------------------------------------------------
// Kernel 0: fp32 -> bf16 conversion (vectorized x4)
// ---------------------------------------------------------------------------
__global__ void cvt_f32_to_bf16_x4(const float4* __restrict__ src,
                                   u16* __restrict__ dst, int n4) {
    int i = blockIdx.x * blockDim.x + threadIdx.x;
    if (i >= n4) return;
    float4 f = src[i];
    u16 o0 = f32_to_bf16(f.x), o1 = f32_to_bf16(f.y);
    u16 o2 = f32_to_bf16(f.z), o3 = f32_to_bf16(f.w);
    uint2 packed;
    packed.x = (uint32_t)o0 | ((uint32_t)o1 << 16);
    packed.y = (uint32_t)o2 | ((uint32_t)o3 << 16);
    *(uint2*)(dst + (size_t)i * 4) = packed;
}

// ---------------------------------------------------------------------------
// Kernel 1: QKV projection  Y = X @ W^T + b   (NT GEMM, bf16 in, f32 acc)
//   X  : bf16 [M=8192, K=1024] row-major
//   W  : bf16 [N=1024, K=1024] row-major
//   out: bf16, vt==0 -> [B,NH,S,HD]  (Q,K);  vt==1 -> [B,NH,HD,S] (V^T)
// Block = 256 thr = 8 waves; wave tile = 32 rows x 64 cols (8 wmma accs).
// Shared W k-tile (64x32 bf16) double-buffered in LDS (all waves reuse it).
// Grid = (N/64, M/256)
// ---------------------------------------------------------------------------
__global__ __launch_bounds__(256)
void qkv_gemm(const u16* __restrict__ X, const u16* __restrict__ W,
              const float* __restrict__ bias, u16* __restrict__ Y, int vt) {
    __shared__ u16 wtile[2][64 * WSTRIDE];   // 2 x 5KB, bank-conflict padded

    const int tid  = threadIdx.x;
    const int lane = tid & 31;
    const int half = lane >> 4;   // which 16-lane half of the wave32
    const int l16  = lane & 15;
    const int wave = tid >> 5;

    const int m0 = blockIdx.y * 256 + wave * 32;   // wave's 32-row tile
    const int n0 = blockIdx.x * 64;

    // W staging: 64 rows x 32 k; 4 threads/row, 8 u16 (one b128) each
    const int srow = tid >> 2;
    const int scol = (tid & 3) * 8;
    const u16* wsrc = W + (size_t)(n0 + srow) * H_ + scol;
    const int soff  = srow * WSTRIDE + scol;

    const u16* arow0 = X + (size_t)(m0 + l16) * H_;        // A strip 0: lane = row
    const u16* arow1 = X + (size_t)(m0 + 16 + l16) * H_;   // A strip 1

    // preload k-tile 0
    *(uint4*)&wtile[0][soff] = *(const uint4*)(wsrc);
    __syncthreads();

    v8f acc[2][4] = {};

    for (int it = 0; it < H_ / 32; ++it) {
        const int k0  = it * 32;
        const int buf = it & 1;
        const bool has_next = (it + 1) < (H_ / 32);

        // early global load of next W tile (held in regs until after compute)
        uint4 wst;
        if (has_next) {
            wst = *(const uint4*)(wsrc + k0 + 32);
            __builtin_prefetch(wsrc + k0 + 64, 0, 3);   // global_prefetch_b8
        }

        // A fragments 16x32: lanes<16 take K {k0..k0+7, k0+16..23}, lanes>=16 +8
        union { uint4 q[2]; v16bf v; } A0, A1;
        {
            const u16* ak = arow0 + k0 + half * 8;
            A0.q[0] = *(const uint4*)(ak);
            A0.q[1] = *(const uint4*)(ak + 16);
        }
        {
            const u16* ak = arow1 + k0 + half * 8;
            A1.q[0] = *(const uint4*)(ak);
            A1.q[1] = *(const uint4*)(ak + 16);
        }

#pragma unroll
        for (int t = 0; t < 4; ++t) {
            // B fragment 32x16 from LDS: lane l16 = column N, K {8h..8h+7, 16+8h..}
            const u16* bk = &wtile[buf][(t * 16 + l16) * WSTRIDE + half * 8];
            union { uint4 q[2]; v16bf v; } Bf;
            Bf.q[0] = *(const uint4*)(bk);
            Bf.q[1] = *(const uint4*)(bk + 16);
            acc[0][t] = __builtin_amdgcn_wmma_f32_16x16x32_bf16(
                false, A0.v, false, Bf.v, (short)0, acc[0][t], false, false);
            acc[1][t] = __builtin_amdgcn_wmma_f32_16x16x32_bf16(
                false, A1.v, false, Bf.v, (short)0, acc[1][t], false, false);
        }

        if (has_next) *(uint4*)&wtile[buf ^ 1][soff] = wst;
        __syncthreads();
    }

    // Epilogue: + bias, write bf16 in head-split layout
#pragma unroll
    for (int t = 0; t < 4; ++t) {
        const int n = n0 + t * 16 + l16;
        const float bv = bias[n];
        const int h = n >> 6, d = n & 63;
#pragma unroll
        for (int s = 0; s < 2; ++s) {
#pragma unroll
            for (int i = 0; i < 8; ++i) {
                const int m = m0 + s * 16 + i + half * 8;   // C-frag: M = i + 8*half
                const int b = m >> 10, sq = m & 1023;
                const u16 o = f32_to_bf16(acc[s][t][i] + bv);
                size_t idx = vt
                    ? (((size_t)(b * NH_ + h) * HD_ + d) * S_ + sq)
                    : (((size_t)(b * NH_ + h) * S_ + sq) * HD_ + d);
                Y[idx] = o;
            }
        }
    }
}

// ---------------------------------------------------------------------------
// Kernel 2: flash attention per (b, h).
//   Qb,Kb : bf16 [B,NH,S,64];  Vt : bf16 [B,NH,64,S];  Out : f32 [B,S,H]
// Block = 256 thr = 8 waves; each wave owns 16 query rows; 32 keys / iter.
// Grid = (S/128, NH, B)
// ---------------------------------------------------------------------------
__global__ __launch_bounds__(256)
void flash_attn(const u16* __restrict__ Qb, const u16* __restrict__ Kb,
                const u16* __restrict__ Vt, const float* __restrict__ mask,
                float* __restrict__ Out) {
    __shared__ u16 lds_p[8 * 16 * WSTRIDE];   // per-wave 16x32 P tile, padded rows

    const int lane = threadIdx.x & 31;
    const int wave = threadIdx.x >> 5;
    const int half = lane >> 4;
    const int l16  = lane & 15;

    const int b = blockIdx.z;
    const int h = blockIdx.y;
    const int q0 = blockIdx.x * 128 + wave * 16;

    const u16* Qh = Qb + (size_t)(b * NH_ + h) * S_ * HD_;
    const u16* Kh = Kb + (size_t)(b * NH_ + h) * S_ * HD_;
    const u16* Vh = Vt + (size_t)(b * NH_ + h) * HD_ * S_;
    const float* mk = mask + (size_t)b * S_;

    // Q A-fragments (hd 0..31 and 32..63) resident in registers
    union { uint4 q[2]; v16bf v; } Alo, Ahi;
    {
        const u16* qrow = Qh + (size_t)(q0 + l16) * HD_;
        Alo.q[0] = *(const uint4*)(qrow + half * 8);
        Alo.q[1] = *(const uint4*)(qrow + half * 8 + 16);
        Ahi.q[0] = *(const uint4*)(qrow + 32 + half * 8);
        Ahi.q[1] = *(const uint4*)(qrow + 32 + half * 8 + 16);
    }

    v8f O[4] = {};                 // context accumulators, 4 d-tiles of 16
    float mrow[8], lrow[8];
#pragma unroll
    for (int i = 0; i < 8; ++i) { mrow[i] = -3.0e38f; lrow[i] = 0.0f; }

    u16* ptile = lds_p + wave * (16 * WSTRIDE);
    const float l2e   = 1.44269504088896340736f;
    const float alpha = 0.125f * l2e;   // 1/sqrt(64) folded into exp2 domain

    for (int j = 0; j < S_; j += 32) {
        if (j + 32 < S_) {  // prefetch next K/V rows -> global_prefetch_b8
            __builtin_prefetch(Kh + (size_t)(j + 32 + l16) * HD_, 0, 3);
            __builtin_prefetch(Vh + (size_t)l16 * S_ + j + 32, 0, 3);
        }

        // ---- scores: two 16-key subtiles, each = 2 wmma over hd ----
        v8f Sa = {}, Sb = {};
        {
            const u16* krow = Kh + (size_t)(j + l16) * HD_;   // lane = key col
            union { uint4 q[2]; v16bf v; } Bf;
            Bf.q[0] = *(const uint4*)(krow + half * 8);
            Bf.q[1] = *(const uint4*)(krow + half * 8 + 16);
            Sa = __builtin_amdgcn_wmma_f32_16x16x32_bf16(false, Alo.v, false, Bf.v, (short)0, Sa, false, false);
            Bf.q[0] = *(const uint4*)(krow + 32 + half * 8);
            Bf.q[1] = *(const uint4*)(krow + 32 + half * 8 + 16);
            Sa = __builtin_amdgcn_wmma_f32_16x16x32_bf16(false, Ahi.v, false, Bf.v, (short)0, Sa, false, false);
        }
        {
            const u16* krow = Kh + (size_t)(j + 16 + l16) * HD_;
            union { uint4 q[2]; v16bf v; } Bf;
            Bf.q[0] = *(const uint4*)(krow + half * 8);
            Bf.q[1] = *(const uint4*)(krow + half * 8 + 16);
            Sb = __builtin_amdgcn_wmma_f32_16x16x32_bf16(false, Alo.v, false, Bf.v, (short)0, Sb, false, false);
            Bf.q[0] = *(const uint4*)(krow + 32 + half * 8);
            Bf.q[1] = *(const uint4*)(krow + 32 + half * 8 + 16);
            Sb = __builtin_amdgcn_wmma_f32_16x16x32_bf16(false, Ahi.v, false, Bf.v, (short)0, Sb, false, false);
        }

        // ---- scale + mask bias + online softmax (exp2 domain) ----
        const float bias_a = (1.0f - mk[j + l16])      * -10000.0f * l2e;
        const float bias_b = (1.0f - mk[j + 16 + l16]) * -10000.0f * l2e;
        float rmax[8];
#pragma unroll
        for (int i = 0; i < 8; ++i) {
            float ta = Sa[i] * alpha + bias_a;
            float tb = Sb[i] * alpha + bias_b;
            Sa[i] = ta; Sb[i] = tb;
            float t = fmaxf(ta, tb);                  // row reduce over 16 lanes/half
            t = fmaxf(t, __shfl_xor(t, 1, 32));
            t = fmaxf(t, __shfl_xor(t, 2, 32));
            t = fmaxf(t, __shfl_xor(t, 4, 32));
            t = fmaxf(t, __shfl_xor(t, 8, 32));
            rmax[i] = t;
        }
#pragma unroll
        for (int i = 0; i < 8; ++i) {
            const float mnew = fmaxf(mrow[i], rmax[i]);
            const float corr = exp2f(mrow[i] - mnew);
            mrow[i] = mnew;
            const float pa = exp2f(Sa[i] - mnew);
            const float pb = exp2f(Sb[i] - mnew);
            Sa[i] = pa; Sb[i] = pb;
            float rs = pa + pb;
            rs += __shfl_xor(rs, 1, 32);
            rs += __shfl_xor(rs, 2, 32);
            rs += __shfl_xor(rs, 4, 32);
            rs += __shfl_xor(rs, 8, 32);
            lrow[i] = lrow[i] * corr + rs;
            O[0][i] *= corr; O[1][i] *= corr; O[2][i] *= corr; O[3][i] *= corr;
        }

        // ---- C-layout P -> A-layout via per-wave LDS tile ----
#pragma unroll
        for (int i = 0; i < 8; ++i) {
            const int row = i + half * 8;
            ptile[row * WSTRIDE + l16]      = f32_to_bf16(Sa[i]);
            ptile[row * WSTRIDE + 16 + l16] = f32_to_bf16(Sb[i]);
        }
        asm volatile("" ::: "memory");   // keep ds_load after ds_store; HW is in-order per wave
        union { uint4 q[2]; v16bf v; } Pf;
        {
            const u16* pr = ptile + l16 * WSTRIDE + half * 8;
            Pf.q[0] = *(const uint4*)(pr);
            Pf.q[1] = *(const uint4*)(pr + 16);
        }

        // ---- PV: O[f] += P(16x32) x V(32keys x 16d), V^T rows contiguous ----
#pragma unroll
        for (int f = 0; f < 4; ++f) {
            const u16* vrow = Vh + (size_t)(f * 16 + l16) * S_ + j + half * 8;
            union { uint4 q[2]; v16bf v; } Bv;
            Bv.q[0] = *(const uint4*)(vrow);
            Bv.q[1] = *(const uint4*)(vrow + 16);
            O[f] = __builtin_amdgcn_wmma_f32_16x16x32_bf16(
                false, Pf.v, false, Bv.v, (short)0, O[f], false, false);
        }
    }

    // ---- normalize and store fp32 context, heads re-merged ----
    const size_t obase = ((size_t)b * S_ + q0) * H_ + (size_t)h * HD_;
#pragma unroll
    for (int i = 0; i < 8; ++i) {
        const float inv = 1.0f / lrow[i];
        const int m = i + half * 8;
#pragma unroll
        for (int f = 0; f < 4; ++f) {
            Out[obase + (size_t)m * H_ + f * 16 + l16] = O[f][i] * inv;
        }
    }
}

// ---------------------------------------------------------------------------
// Host-side launcher
// ---------------------------------------------------------------------------
extern "C" void kernel_launch(void* const* d_in, const int* in_sizes, int n_in,
                              void* d_out, int out_size, void* d_ws, size_t ws_size,
                              hipStream_t stream) {
    const float* hidden = (const float*)d_in[0];
    const float* amask  = (const float*)d_in[1];
    const float* Wq     = (const float*)d_in[2];
    const float* bq     = (const float*)d_in[3];
    const float* Wk     = (const float*)d_in[4];
    const float* bk     = (const float*)d_in[5];
    const float* Wv     = (const float*)d_in[6];
    const float* bv     = (const float*)d_in[7];
    float* out = (float*)d_out;

    char* ws = (char*)d_ws;
    u16* Xb  = (u16*)(ws);                          // 16 MB  hidden bf16
    u16* Wqb = (u16*)(ws + ((size_t)16 << 20));     //  2 MB
    u16* Wkb = (u16*)(ws + ((size_t)18 << 20));     //  2 MB
    u16* Wvb = (u16*)(ws + ((size_t)20 << 20));     //  2 MB
    u16* Qb  = (u16*)(ws + ((size_t)22 << 20));     // 16 MB  [B,NH,S,64]
    u16* Kb  = (u16*)(ws + ((size_t)38 << 20));     // 16 MB  [B,NH,S,64]
    u16* Vt  = (u16*)(ws + ((size_t)54 << 20));     // 16 MB  [B,NH,64,S]

    const int nX = B_ * S_ * H_;   // 8M
    const int nW = H_ * H_;        // 1M
    cvt_f32_to_bf16_x4<<<(nX / 4 + 255) / 256, 256, 0, stream>>>((const float4*)hidden, Xb, nX / 4);
    cvt_f32_to_bf16_x4<<<(nW / 4 + 255) / 256, 256, 0, stream>>>((const float4*)Wq, Wqb, nW / 4);
    cvt_f32_to_bf16_x4<<<(nW / 4 + 255) / 256, 256, 0, stream>>>((const float4*)Wk, Wkb, nW / 4);
    cvt_f32_to_bf16_x4<<<(nW / 4 + 255) / 256, 256, 0, stream>>>((const float4*)Wv, Wvb, nW / 4);

    dim3 ggrid(H_ / 64, (B_ * S_) / 256);
    qkv_gemm<<<ggrid, 256, 0, stream>>>(Xb, Wqb, bq, Qb, 0);
    qkv_gemm<<<ggrid, 256, 0, stream>>>(Xb, Wkb, bk, Kb, 0);
    qkv_gemm<<<ggrid, 256, 0, stream>>>(Xb, Wvb, bv, Vt, 1);

    dim3 agrid(S_ / 128, NH_, B_);
    flash_attn<<<agrid, 256, 0, stream>>>(Qb, Kb, Vt, amask, out);
}
